// MultiHeadSelfAttention_27376121545211
// MI455X (gfx1250) — matrix operand order repaired and checked by reading, not compiled, not run on hardware
//
#include <hip/hip_runtime.h>

#define D_MODEL 1024
#define NUM_HEADS 16
#define DEPTH 64
#define BATCH 2
#define SEQ 2048

typedef __attribute__((ext_vector_type(16))) __bf16 v16bf;
typedef __attribute__((ext_vector_type(8)))  float  v8f;

union TileU { v16bf v; unsigned int u[8]; };

__device__ __forceinline__ unsigned short f2bf(float x) {
  unsigned int u = __float_as_uint(x);
  u += 0x7FFFu + ((u >> 16) & 1u);   // round-to-nearest-even
  return (unsigned short)(u >> 16);
}

__device__ __forceinline__ unsigned int pack2(float lo, float hi) {
  return (unsigned int)f2bf(lo) | ((unsigned int)f2bf(hi) << 16);
}

// A operand: 16(M) x 32(K) bf16 tile, rows taken from a row-major bf16 source.
// ISA layout: lane m=L&15, half=L>>4; VGPR v<4 -> K = 8*half+2v, v>=4 -> K = 16+8*half+2(v-4).
__device__ __forceinline__ v16bf load_a_tile(const unsigned short* src, int row0, int k0, int ld) {
  const int lane = threadIdx.x & 31;
  const int m = lane & 15, half = lane >> 4;
  const unsigned short* p = src + (row0 + m) * ld + k0 + half * 8;
  TileU t;
#pragma unroll
  for (int v = 0; v < 4; ++v) t.u[v] = *(const unsigned int*)(p + 2 * v);
  p += 16;
#pragma unroll
  for (int v = 0; v < 4; ++v) t.u[4 + v] = *(const unsigned int*)(p + 2 * v);
  return t.v;
}

// B operand: 32(K) x 16(N) bf16 tile; column n is sourced from row (row0+n) of a
// row-major source (i.e. the source holds B^T). Layout: lane n=L&15, half=L>>4;
// VGPR v<4 -> K = 8*half + 2v, v>=4 -> K = 16 + 8*half + 2(v-4).
__device__ __forceinline__ v16bf load_b_tile(const unsigned short* src, int row0, int k0, int ld) {
  const int lane = threadIdx.x & 31;
  const int n = lane & 15, half = lane >> 4;
  const unsigned short* p = src + (row0 + n) * ld + k0 + half * 8;
  TileU t;
#pragma unroll
  for (int v = 0; v < 8; ++v) t.u[v] = *(const unsigned int*)(p + 2 * v + (v >> 2) * 8);
  return t.v;
}

__device__ __forceinline__ v8f wmma_bf16(v16bf a, v16bf b, v8f c) {
  return __builtin_amdgcn_wmma_f32_16x16x32_bf16(false, a, false, b, (short)0, c, false, false);
}

// ---------------- prep kernels (one-time format conversion) ----------------

__global__ void cvt_bf16_kernel(const float* __restrict__ src, unsigned short* __restrict__ dst, int n) {
  int i = blockIdx.x * 256 + threadIdx.x;
  if (i < n) dst[i] = f2bf(src[i]);
}

// Wt[n][k] = bf16(W[k][n]) for 1024x1024
__global__ void transpose_cvt_kernel(const float* __restrict__ W, unsigned short* __restrict__ Wt) {
  int o = blockIdx.x * 256 + threadIdx.x;
  int n = o >> 10, k = o & 1023;
  Wt[o] = f2bf(W[k * D_MODEL + n]);
}

// ---------------- generic WMMA GEMM: [4096 x 1024] = A @ Wt^T + bias ----------------
// Weight panel (shared by all 8 waves of the block) staged in LDS each K-step.
// mode 0: Q -> bf16 [B][H][S][64], pre-scaled by 1/sqrt(DEPTH)
// mode 1: K -> bf16 [B][H][S][64]
// mode 2: V -> bf16 [B][H][64][S]  (transposed for the P@V B-operand)
// mode 3: f32 plain [M][1024] into d_out
__global__ __launch_bounds__(256) void gemm_kernel(const unsigned short* __restrict__ A,
                                                   const unsigned short* __restrict__ Wt,
                                                   const float* __restrict__ bias,
                                                   unsigned short* __restrict__ outb,
                                                   float* __restrict__ outf,
                                                   int mode) {
  __shared__ __align__(16) unsigned short Bs[64 * 32];  // 64 cols x 32 K, 4 KB

  const int tid  = threadIdx.x;
  const int wid  = tid >> 5;
  const int lane = tid & 31;
  const int half = lane >> 4;
  const int row0 = (blockIdx.y * 8 + wid) * 16;   // 16-row strip per wave
  const int n0   = blockIdx.x * 64;               // 64-col group per block

  // cooperative-copy indices: one uint4 (8 bf16) per thread per fill
  const int crow = tid >> 2;          // 0..63  (weight row within panel)
  const int coff = (tid & 3) * 8;     // 0,8,16,24 (shorts within 32-K slice)

  v8f acc[4] = {};
  for (int k = 0; k < D_MODEL; k += 32) {
    *(uint4*)(Bs + crow * 32 + coff) =
        *(const uint4*)(Wt + (size_t)(n0 + crow) * D_MODEL + k + coff);
    __syncthreads();

    v16bf a = load_a_tile(A, row0, k, D_MODEL);
#pragma unroll
    for (int t = 0; t < 4; ++t) {
      v16bf b = load_b_tile(Bs, t * 16, 0, 32);
      acc[t] = wmma_bf16(a, b, acc[t]);
    }
    __syncthreads();
  }

  const int coln = lane & 15;
#pragma unroll
  for (int t = 0; t < 4; ++t) {
    const int n = n0 + t * 16 + coln;
    const float bv = bias[n];
#pragma unroll
    for (int r = 0; r < 8; ++r) {
      const int m = row0 + half * 8 + r;
      const float val = acc[t][r] + bv;
      if (mode == 3) {
        outf[(size_t)m * D_MODEL + n] = val;
      } else {
        const int b = m >> 11, s = m & (SEQ - 1);
        const int h = n >> 6,  d = n & 63;
        if (mode == 0) {
          outb[(((size_t)(b * NUM_HEADS + h) * SEQ) + s) * DEPTH + d] = f2bf(val * 0.125f);
        } else if (mode == 1) {
          outb[(((size_t)(b * NUM_HEADS + h) * SEQ) + s) * DEPTH + d] = f2bf(val);
        } else {
          outb[(((size_t)(b * NUM_HEADS + h) * DEPTH) + d) * SEQ + s] = f2bf(val);
        }
      }
    }
  }
}

// ---------------- flash attention ----------------
// Block = 8 waves, all on the same (b,h); each wave owns a 16-row q-tile.
// K-tile (32x64) and V^T-tile (64x32) staged once per 32-key step in LDS.
__global__ __launch_bounds__(256) void attn_kernel(const unsigned short* __restrict__ Qb,
                                                   const unsigned short* __restrict__ Kb,
                                                   const unsigned short* __restrict__ Vt,
                                                   unsigned short* __restrict__ Ab) {
  __shared__ __align__(16) unsigned short Ks[32 * 64];  // [key][d], 4 KB
  __shared__ __align__(16) unsigned short Vs[64 * 32];  // [d][key], 4 KB

  const int tid  = threadIdx.x;
  const int wid  = tid >> 5;
  const int lane = tid & 31;
  const int half = lane >> 4;
  const int bh   = blockIdx.y;                 // b*16 + h
  const int q0   = (blockIdx.x * 8 + wid) * 16;

  const unsigned short* Qh = Qb + (size_t)bh * SEQ * DEPTH;
  const unsigned short* Kh = Kb + (size_t)bh * SEQ * DEPTH;
  const unsigned short* Vh = Vt + (size_t)bh * DEPTH * SEQ;

  // cooperative-copy indices
  const int krow = tid >> 3;          // 0..31, K-tile row (key)
  const int koff = (tid & 7) * 8;     // shorts within 64-d row
  const int vrow = tid >> 2;          // 0..63, V^T row (d)
  const int voff = (tid & 3) * 8;     // shorts within 32-key row

  // Q tile as B operand (columns of Q^T = rows of Q), resident for whole loop
  const v16bf qt0 = load_b_tile(Qh, q0, 0,  DEPTH);
  const v16bf qt1 = load_b_tile(Qh, q0, 32, DEPTH);

  v8f o0 = {}, o1 = {}, o2 = {}, o3 = {};
  float m_run = -1e30f, l_run = 0.f;

  for (int j = 0; j < SEQ; j += 32) {
    *(uint4*)(Ks + krow * 64 + koff) = *(const uint4*)(Kh + (size_t)(j + krow) * DEPTH + koff);
    *(uint4*)(Vs + vrow * 32 + voff) = *(const uint4*)(Vh + (size_t)vrow * SEQ + j + voff);
    __syncthreads();

    // S^T = K_tile(16x64) @ Q^T(64x16): C has q = lane&15, key = 8*half + r
    v16bf ka0 = load_a_tile(Ks, 0,  0,  DEPTH);
    v16bf ka1 = load_a_tile(Ks, 0,  32, DEPTH);
    v16bf kb0 = load_a_tile(Ks, 16, 0,  DEPTH);
    v16bf kb1 = load_a_tile(Ks, 16, 32, DEPTH);
    v8f st0 = {}; st0 = wmma_bf16(ka0, qt0, st0); st0 = wmma_bf16(ka1, qt1, st0);
    v8f st1 = {}; st1 = wmma_bf16(kb0, qt0, st1); st1 = wmma_bf16(kb1, qt1, st1);

    // online softmax stats for q-row = lane&15 (merge halves with shfl_xor 16)
    float tm = st0[0];
#pragma unroll
    for (int r = 1; r < 8; ++r) tm = fmaxf(tm, st0[r]);
#pragma unroll
    for (int r = 0; r < 8; ++r) tm = fmaxf(tm, st1[r]);
    tm = fmaxf(tm, __shfl_xor(tm, 16, 32));
    const float mnew  = fmaxf(m_run, tm);
    const float alpha = __expf(m_run - mnew);

    float p0[8], p1[8], ls = 0.f;
#pragma unroll
    for (int r = 0; r < 8; ++r) { p0[r] = __expf(st0[r] - mnew); ls += p0[r]; }
#pragma unroll
    for (int r = 0; r < 8; ++r) { p1[r] = __expf(st1[r] - mnew); ls += p1[r]; }
    ls += __shfl_xor(ls, 16, 32);
    l_run = l_run * alpha + ls;
    m_run = mnew;

    // repack exp(scores) into WMMA A layout (16 q x 32 keys) — all in-lane:
    // A needs key K = 8*half+2v (v<4) from st0[r=2v], K = 16+8*half+2(v-4) from st1.
    TileU pa;
#pragma unroll
    for (int v = 0; v < 4; ++v) pa.u[v]     = pack2(p0[2 * v], p0[2 * v + 1]);
#pragma unroll
    for (int v = 0; v < 4; ++v) pa.u[4 + v] = pack2(p1[2 * v], p1[2 * v + 1]);

    // rescale O rows (row index = 8*half + r, stat lives in lane 8*half+r)
#pragma unroll
    for (int r = 0; r < 8; ++r) {
      const float af = __shfl(alpha, (half << 3) + r, 32);
      o0[r] *= af; o1[r] *= af; o2[r] *= af; o3[r] *= af;
    }
    // O += P @ V ; B columns = V columns (rows of V^T tile in LDS)
    o0 = wmma_bf16(pa.v, load_b_tile(Vs,  0, 0, 32), o0);
    o1 = wmma_bf16(pa.v, load_b_tile(Vs, 16, 0, 32), o1);
    o2 = wmma_bf16(pa.v, load_b_tile(Vs, 32, 0, 32), o2);
    o3 = wmma_bf16(pa.v, load_b_tile(Vs, 48, 0, 32), o3);
    __syncthreads();
  }

  const float inv = 1.f / l_run;
  const int b = bh >> 4, h = bh & 15;
  const int col = lane & 15;
#pragma unroll
  for (int r = 0; r < 8; ++r) {
    const float sf = __shfl(inv, (half << 3) + r, 32);
    const int srow = q0 + half * 8 + r;
    const size_t base = ((size_t)(b * SEQ + srow)) * D_MODEL + h * DEPTH;
    Ab[base +  0 + col] = f2bf(o0[r] * sf);
    Ab[base + 16 + col] = f2bf(o1[r] * sf);
    Ab[base + 32 + col] = f2bf(o2[r] * sf);
    Ab[base + 48 + col] = f2bf(o3[r] * sf);
  }
}

// ---------------- launch ----------------

extern "C" void kernel_launch(void* const* d_in, const int* in_sizes, int n_in,
                              void* d_out, int out_size, void* d_ws, size_t ws_size,
                              hipStream_t stream) {
  const float* X  = (const float*)d_in[0];
  const float* wq = (const float*)d_in[1];
  const float* bq = (const float*)d_in[2];
  const float* wk = (const float*)d_in[3];
  const float* bk = (const float*)d_in[4];
  const float* wv = (const float*)d_in[5];
  const float* bv = (const float*)d_in[6];
  const float* wo = (const float*)d_in[7];
  const float* bo = (const float*)d_in[8];
  float* out = (float*)d_out;

  // workspace layout (ushort/bf16 units), total ~48 MB
  unsigned short* ws  = (unsigned short*)d_ws;
  unsigned short* Xb  = ws;                    // 4096*1024
  unsigned short* Wqt = Xb  + 4194304;         // 1024*1024 each
  unsigned short* Wkt = Wqt + 1048576;
  unsigned short* Wvt = Wkt + 1048576;
  unsigned short* Wot = Wvt + 1048576;
  unsigned short* Qb  = Wot + 1048576;         // [B][H][S][64]
  unsigned short* Kb  = Qb  + 4194304;
  unsigned short* Vt  = Kb  + 4194304;         // [B][H][64][S]
  unsigned short* Ab  = Vt  + 4194304;         // [B][S][1024]

  cvt_bf16_kernel<<<16384, 256, 0, stream>>>(X, Xb, 4194304);
  transpose_cvt_kernel<<<4096, 256, 0, stream>>>(wq, Wqt);
  transpose_cvt_kernel<<<4096, 256, 0, stream>>>(wk, Wkt);
  transpose_cvt_kernel<<<4096, 256, 0, stream>>>(wv, Wvt);
  transpose_cvt_kernel<<<4096, 256, 0, stream>>>(wo, Wot);

  dim3 gg(16, 32);  // 16 col-groups x (256 row-tiles / 8 waves)
  gemm_kernel<<<gg, 256, 0, stream>>>(Xb, Wqt, bq, Qb, nullptr, 0);
  gemm_kernel<<<gg, 256, 0, stream>>>(Xb, Wkt, bk, Kb, nullptr, 1);
  gemm_kernel<<<gg, 256, 0, stream>>>(Xb, Wvt, bv, Vt, nullptr, 2);

  attn_kernel<<<dim3(16, 32), 256, 0, stream>>>(Qb, Kb, Vt, Ab);

  gemm_kernel<<<gg, 256, 0, stream>>>(Ab, Wot, bo, nullptr, out, 3);
}
